// NeuralCDE_79173427134597
// MI455X (gfx1250) — matrix operand order repaired and checked by reading, not compiled, not run on hardware
//
#include <hip/hip_runtime.h>
#include <hip/hip_bf16.h>
#include <math.h>

// ---------------------------------------------------------------------------
// Neural CDE forward on MI455X (gfx1250).
// Persistent-kernel design: batch rows integrate independently, so each of the
// 32 blocks owns a 16-row M-tile and runs all 500 Heun steps locally.
// All GEMMs go through v_wmma_f32_16x16x32_f16 with f32 accumulation.
// ---------------------------------------------------------------------------

typedef __attribute__((ext_vector_type(16))) _Float16 v16h;
typedef __attribute__((ext_vector_type(8)))  _Float16 v8h;
typedef __attribute__((ext_vector_type(8)))  float    v8f;

#define CDE_B       512
#define CDE_L       2000
#define CDE_D       16
#define CDE_H       128
#define CDE_W       256
#define CDE_LBL     10
#define CDE_NSTEPS  500

// LDS strides (elements), padded to break 64-bank conflict patterns.
#define AS 136   // y / y2 activation tile, f16, 16 x 128 (+8 pad)
#define HS 264   // hidden activation tiles, f16, 16 x 256 (+8 pad)
#define KS 132   // k1/k2 tiles, f32, 16 x 128 (+4 pad)

// Packed-weight tile counts: tile = 32 lanes x 16 halves = 512 halves.
#define W1_TILES  (16 * 4)    // N=256 -> 16 ntiles, K=128 -> 4 ktiles
#define W2_TILES  (16 * 8)    // N=256, K=256
#define W3_TILES  (128 * 8)   // N=2048, K=256
#define PACK_TOTAL ((W1_TILES + W2_TILES + W3_TILES) * 512)

// ---------------------------------------------------------------------------
// Pack W1/W2/W3 (row-major f32, shape [N,K]) into the B-fragment layout of
// V_WMMA_F32_16X16X32_F16:
//   lane 0..15  : N = ntile*16 + lane,    j 0..7 -> K = ktile*32 + j,     j 8..15 -> K = ktile*32 + 8 + j
//   lane 16..31 : N = ntile*16 + lane-16, j 0..7 -> K = ktile*32 + 8 + j, j 8..15 -> K = ktile*32 + 16 + j
// Each lane's 16 halves are contiguous -> two coalesced b128 loads per tile.
// ---------------------------------------------------------------------------
__global__ void cde_pack_weights(const float* __restrict__ W1,
                                 const float* __restrict__ W2,
                                 const float* __restrict__ W3,
                                 _Float16* __restrict__ wp) {
  int e = blockIdx.x * blockDim.x + threadIdx.x;
  if (e >= PACK_TOTAL) return;
  int tile = e >> 9;
  int r    = e & 511;
  int lane = r >> 4;
  int j    = r & 15;

  const float* W;
  int kdim, ktiles, tbase;
  if (tile < W1_TILES)                 { W = W1; kdim = 128; ktiles = 4; tbase = 0; }
  else if (tile < W1_TILES + W2_TILES) { W = W2; kdim = 256; ktiles = 8; tbase = W1_TILES; }
  else                                 { W = W3; kdim = 256; ktiles = 8; tbase = W1_TILES + W2_TILES; }

  int lt    = tile - tbase;
  int ntile = lt / ktiles;
  int ktile = lt - ntile * ktiles;
  int n = ntile * 16 + (lane & 15);
  int k = ktile * 32 + ((lane & 16) ? 8 : 0) + (j < 8 ? j : j + 8);
  wp[e] = (_Float16)W[n * kdim + k];
}

// ---------------------------------------------------------------------------
// Fragment helpers
// ---------------------------------------------------------------------------
union AFrag { v16h v; v8h h[2]; };

// A-matrix 16x32 f16 layout:
//   lane 0..15  : M = lane,    VGPR0-3 = K+0..7,  VGPR4-7 = K+16..23
//   lane 16..31 : M = lane-16, VGPR0-3 = K+8..15, VGPR4-7 = K+24..31
__device__ __forceinline__ v16h load_a_lds(const _Float16* act, int stride,
                                           int kk, int lane) {
  int m  = lane & 15;
  int kb = kk + ((lane & 16) ? 8 : 0);
  AFrag f;
  f.h[0] = *(const v8h*)(act + m * stride + kb);
  f.h[1] = *(const v8h*)(act + m * stride + kb + 16);
  return f.v;
}

__device__ __forceinline__ v16h load_b_pk(const _Float16* pack, int tile, int lane) {
  return *(const v16h*)(pack + (((size_t)tile << 5) + (size_t)lane) * 16);
}

__device__ __forceinline__ v8f wmma_f16(v16h a, v16h b, v8f c) {
  return __builtin_amdgcn_wmma_f32_16x16x32_f16(false, a, false, b,
                                                (short)0, c, false, false);
}

// ---------------------------------------------------------------------------
// One evaluation of f(t, y):
//   h1 = relu(y @ W1.T + b1); h2 = relu(h1 @ W2.T + b2);
//   o  = tanh(h2 @ W3.T + b3).reshape(16, H, D)
//   k  = einsum('bhd,bd->bh', o, dXdt(t))
// The einsum is fused into GEMM3's epilogue (each 16x16 n-tile == one h,
// all 16 d), using a 16-lane shfl_xor reduction over d.
// ---------------------------------------------------------------------------
__device__ __forceinline__ void eval_f(
    float t, const float* __restrict__ ts, const float* __restrict__ coeffs,
    int b0, const _Float16* __restrict__ W1p, const _Float16* __restrict__ W2p,
    const _Float16* __restrict__ W3p, const float* __restrict__ b1,
    const float* __restrict__ b2, const float* __restrict__ b3,
    const _Float16* Ain, _Float16* H1, _Float16* H2, float* dXs, float* Kout) {
  const int tid  = threadIdx.x;
  const int w    = tid >> 5;
  const int lane = tid & 31;
  const int nin  = lane & 15;
  const int mb   = (lane & 16) ? 8 : 0;

  // ---- dXdt(t) for this block's 16 rows (ts is a uniform grid) ----
  if (tid < 256) {
    int bl = tid >> 4;
    int d  = tid & 15;
    float t0 = ts[0], t1 = ts[CDE_L - 1];
    float hinv = (float)(CDE_L - 1) / (t1 - t0);
    int idx = (int)floorf((t - t0) * hinv);
    idx = idx < 0 ? 0 : (idx > CDE_L - 2 ? CDE_L - 2 : idx);
    float tau = t - ts[idx];
    size_t cs   = (size_t)(CDE_L - 1) * CDE_D;
    size_t base = ((size_t)(b0 + bl) * 4) * cs + (size_t)idx * CDE_D + d;
    float dd = coeffs[base];            // coeffs[:,0] = d_
    float cc = coeffs[base + cs];       // coeffs[:,1] = c_
    float bb = coeffs[base + 2 * cs];   // coeffs[:,2] = b_
    dXs[bl * 16 + d] = bb + 2.0f * tau * cc + 3.0f * tau * tau * dd;
  }

  // ---- GEMM1: 16x256 = y(16x128) @ W1.T, relu ----
  {
    v8f acc = {0.f, 0.f, 0.f, 0.f, 0.f, 0.f, 0.f, 0.f};
#pragma unroll
    for (int kt = 0; kt < 4; ++kt) {
      v16h a = load_a_lds(Ain, AS, kt * 32, lane);
      v16h b = load_b_pk(W1p, w * 4 + kt, lane);
      acc = wmma_f16(a, b, acc);
    }
    float bias = b1[w * 16 + nin];
#pragma unroll
    for (int r = 0; r < 8; ++r) {
      float v = acc[r] + bias;
      H1[(mb + r) * HS + w * 16 + nin] = (_Float16)(v > 0.f ? v : 0.f);
    }
  }
  __syncthreads();

  // ---- GEMM2: 16x256 = h1(16x256) @ W2.T, relu ----
  {
    v8f acc = {0.f, 0.f, 0.f, 0.f, 0.f, 0.f, 0.f, 0.f};
#pragma unroll
    for (int kt = 0; kt < 8; ++kt) {
      v16h a = load_a_lds(H1, HS, kt * 32, lane);
      v16h b = load_b_pk(W2p, w * 8 + kt, lane);
      acc = wmma_f16(a, b, acc);
    }
    float bias = b2[w * 16 + nin];
#pragma unroll
    for (int r = 0; r < 8; ++r) {
      float v = acc[r] + bias;
      H2[(mb + r) * HS + w * 16 + nin] = (_Float16)(v > 0.f ? v : 0.f);
    }
  }
  __syncthreads();

  // ---- GEMM3 + fused tanh + einsum against dX ----
  // wave w owns h = w*8 .. w*8+7; within a tile, lane column == d.
#pragma unroll 1
  for (int hh = 0; hh < 8; ++hh) {
    int h = w * 8 + hh;
    v8f acc = {0.f, 0.f, 0.f, 0.f, 0.f, 0.f, 0.f, 0.f};
#pragma unroll
    for (int kt = 0; kt < 8; ++kt) {
      v16h a = load_a_lds(H2, HS, kt * 32, lane);
      v16h b = load_b_pk(W3p, h * 8 + kt, lane);
      acc = wmma_f16(a, b, acc);
    }
    int d = nin;
    float bias = b3[h * 16 + d];
    float p[8];
#pragma unroll
    for (int r = 0; r < 8; ++r)
      p[r] = tanhf(acc[r] + bias) * dXs[(mb + r) * 16 + d];
    // reduce over d (16 lanes within each half-wave group)
#pragma unroll
    for (int r = 0; r < 8; ++r) {
#pragma unroll
      for (int off = 1; off < 16; off <<= 1)
        p[r] += __shfl_xor(p[r], off, 32);
    }
    if (nin < 8) Kout[(mb + nin) * KS + h] = p[nin];
  }
  __syncthreads();
}

// ---------------------------------------------------------------------------
// Main persistent kernel: 32 blocks x 512 threads (16 wave32s).
// Block bx integrates batch rows [16*bx, 16*bx+16) through all 500 steps.
// ---------------------------------------------------------------------------
__global__ __launch_bounds__(512, 1)
void cde_solve(const float* __restrict__ ts, const float* __restrict__ coeffs,
               const float* __restrict__ x0, const float* __restrict__ b1,
               const float* __restrict__ b2, const float* __restrict__ b3,
               const float* __restrict__ Wl1, const float* __restrict__ bl1,
               const float* __restrict__ Wl2, const float* __restrict__ bl2,
               const _Float16* __restrict__ wpack, float* __restrict__ out) {
  __shared__ _Float16 Ain[16 * AS];
  __shared__ _Float16 H1[16 * HS];
  __shared__ _Float16 H2[16 * HS];
  __shared__ float    Yf[16 * CDE_H];
  __shared__ float    K1[16 * KS];
  __shared__ float    K2[16 * KS];
  __shared__ float    dXs[256];

  const _Float16* W1p = wpack;
  const _Float16* W2p = wpack + (size_t)W1_TILES * 512;
  const _Float16* W3p = wpack + (size_t)(W1_TILES + W2_TILES) * 512;

  const int tid = threadIdx.x;
  const int b0  = blockIdx.x * 16;

  // ---- y0 = x0 @ Wl1.T + bl1 ----
#pragma unroll
  for (int i = 0; i < 4; ++i) {
    int e = tid + i * 512;          // e in [0, 2048)
    int m = e >> 7, h = e & 127;
    float s = bl1[h];
    const float* xr = x0 + (size_t)(b0 + m) * CDE_D;
    const float* wr = Wl1 + (size_t)h * CDE_D;
#pragma unroll
    for (int d = 0; d < CDE_D; ++d) s += xr[d] * wr[d];
    Yf[e] = s;
    Ain[m * AS + h] = (_Float16)s;
  }
  __syncthreads();

  const float t0 = ts[0];
  const float t1 = ts[CDE_L - 1];
  const float dt = (t1 - t0) / (float)CDE_NSTEPS;

  // ---- Heun integration, 500 sequential steps ----
#pragma unroll 1
  for (int k = 0; k < CDE_NSTEPS; ++k) {
    float t = t0 + (float)k * dt;

    eval_f(t, ts, coeffs, b0, W1p, W2p, W3p, b1, b2, b3, Ain, H1, H2, dXs, K1);

    // y2 = y + dt*k1  (f16 into Ain for the next eval)
#pragma unroll
    for (int i = 0; i < 4; ++i) {
      int e = tid + i * 512;
      int m = e >> 7, h = e & 127;
      float v = Yf[e] + dt * K1[m * KS + h];
      Ain[m * AS + h] = (_Float16)v;
    }
    __syncthreads();

    eval_f(t + dt, ts, coeffs, b0, W1p, W2p, W3p, b1, b2, b3, Ain, H1, H2, dXs, K2);

    // y += 0.5*dt*(k1 + k2)
#pragma unroll
    for (int i = 0; i < 4; ++i) {
      int e = tid + i * 512;
      int m = e >> 7, h = e & 127;
      float v = Yf[e] + 0.5f * dt * (K1[m * KS + h] + K2[m * KS + h]);
      Yf[e] = v;
      Ain[m * AS + h] = (_Float16)v;
    }
    __syncthreads();
  }

  // ---- out = softmax(yT @ Wl2.T + bl2) ----
  if (tid < 16) {
    float lo[CDE_LBL];
    float mx = -1e30f;
#pragma unroll
    for (int j = 0; j < CDE_LBL; ++j) {
      float s = bl2[j];
      const float* yr = Yf + tid * CDE_H;
      const float* wr = Wl2 + (size_t)j * CDE_H;
      for (int h = 0; h < CDE_H; ++h) s += yr[h] * wr[h];
      lo[j] = s;
      mx = s > mx ? s : mx;
    }
    float sum = 0.f;
#pragma unroll
    for (int j = 0; j < CDE_LBL; ++j) { lo[j] = expf(lo[j] - mx); sum += lo[j]; }
    float inv = 1.0f / sum;
#pragma unroll
    for (int j = 0; j < CDE_LBL; ++j)
      out[(size_t)(b0 + tid) * CDE_LBL + j] = lo[j] * inv;
  }
}

// ---------------------------------------------------------------------------
extern "C" void kernel_launch(void* const* d_in, const int* in_sizes, int n_in,
                              void* d_out, int out_size, void* d_ws,
                              size_t ws_size, hipStream_t stream) {
  (void)in_sizes; (void)n_in; (void)out_size; (void)ws_size;
  const float* ts     = (const float*)d_in[0];
  const float* coeffs = (const float*)d_in[1];
  const float* x0     = (const float*)d_in[2];
  const float* W1     = (const float*)d_in[3];
  const float* b1     = (const float*)d_in[4];
  const float* W2     = (const float*)d_in[5];
  const float* b2     = (const float*)d_in[6];
  const float* W3     = (const float*)d_in[7];
  const float* b3     = (const float*)d_in[8];
  const float* Wl1    = (const float*)d_in[9];
  const float* bl1    = (const float*)d_in[10];
  const float* Wl2    = (const float*)d_in[11];
  const float* bl2    = (const float*)d_in[12];

  _Float16* wpack = (_Float16*)d_ws;

  cde_pack_weights<<<(PACK_TOTAL + 255) / 256, 256, 0, stream>>>(W1, W2, W3, wpack);
  cde_solve<<<32, 512, 0, stream>>>(ts, coeffs, x0, b1, b2, b3, Wl1, bl1, Wl2,
                                    bl2, wpack, (float*)d_out);
}
